// scaledDotProductAttention_27453430956590
// MI455X (gfx1250) — compile-verified, matrix operand order checked by reading, not run on hardware
//
#include <hip/hip_runtime.h>
#include <hip/hip_bf16.h>
#include <hip/hip_fp16.h>

// ---------------------------------------------------------------------------
// Single-head attention: out = softmax((q wQ)(k wK)^T / 64) (v wV)
// S=8192, D_IN=DK=DV=1024.  All GEMMs via V_WMMA_F32_16X16X32_F16 (wave32).
// Workspace: Qh,Kh (f16), Vt (f16, stored transposed), P (f16 8192x8192,
// L2-resident).  GEMM: block 64x256 (8 waves 2x4), wave tile 32x64 (2x4 WMMA),
// KTILE=64, double-buffered LDS, async GLOBAL_LOAD_ASYNC_TO_LDS_B128 staging
// for the pure-f16 GEMMs (ASYNCcnt + s_wait_asynccnt).
// ---------------------------------------------------------------------------

typedef __attribute__((ext_vector_type(16))) _Float16 v16h;
typedef __attribute__((ext_vector_type(8)))  float    v8f;
typedef __attribute__((ext_vector_type(8)))  unsigned v8u;

#define S_LEN 8192
#define D_INF 1024
#define DK    1024
#define DV    1024

#define KTILE      64
#define KPAD       8
#define LDSSTRIDE  (KTILE + KPAD)   // 72 halfs = 144B row stride (9*16B aligned)

union FragU { v16h h; v8u u; };

// LDS byte offset of a __shared__ pointer: flat->LDS mapping uses addr[31:0]
// (ISA 10.2 aperture rules), which matches the async-op VDST semantics.
__device__ __forceinline__ unsigned lds_off(const void* p) {
    return (unsigned)(size_t)p;
}

// One CDNA5 async copy: 16 bytes/lane, global -> LDS, tracked by ASYNCcnt.
__device__ __forceinline__ void async_b128(unsigned lds, const void* g) {
    asm volatile("global_load_async_to_lds_b128 %0, %1, off"
                 :: "v"(lds), "v"((unsigned long long)(size_t)g)
                 : "memory");
}

// Load one 16x32 f16 fragment from LDS stored [row][k] (k contiguous, pairs
// 32b-aligned).  CDNA5 16-bit A-matrix 16x32 layout (ISA 7.12.2): lanes 0-15
// hold rows 0-15 with K={0..7,16..23}, lanes 16-31 K={8..15,24..31}.  B
// fragments use the same pattern with row == N (B tile kept [N][K] in LDS).
__device__ __forceinline__ v16h load_frag_nk(const _Float16* sm, int row0, int lane) {
    FragU f;
    const unsigned* p =
        (const unsigned*)(sm + (size_t)(row0 + (lane & 15)) * LDSSTRIDE);
    const int kh = (lane & 16) ? 4 : 0;
#pragma unroll
    for (int vr = 0; vr < 8; ++vr) {
        const int ku = kh + ((vr & 4) ? 8 : 0) + (vr & 3);  // uint index == k/2
        f.u[vr] = p[ku];
    }
    return f.h;
}

// Generic tiled GEMM: C = scale * A[M,K] x B.
//   B_NXK:  B stored [N][K] (rows are columns) vs [K][N] (transposed in LDS).
//   STORE_T: write C transposed as C[N][M] (used to produce Vt).
//   f16 A / f16 [N][K] B stage via async-to-LDS; f32 paths convert in flight.
template <typename AT, typename BT, bool B_NXK, typename OT, bool STORE_T>
__global__ __launch_bounds__(256) void gemm_wmma_kernel(
    const AT* __restrict__ A, const BT* __restrict__ B, OT* __restrict__ C,
    int M, int N, int K, float scale)
{
    constexpr bool ASYNC_A = __is_same(AT, _Float16);
    constexpr bool ASYNC_B = B_NXK && __is_same(BT, _Float16);

    __shared__ _Float16 Ash[2][64  * LDSSTRIDE];
    __shared__ _Float16 Bsh[2][256 * LDSSTRIDE];

    const int tid  = threadIdx.x;
    const int lane = tid & 31;
    const int wave = tid >> 5;
    const int wm   = wave >> 2;         // 0..1 along M (32 rows)
    const int wn   = wave & 3;          // 0..3 along N (64 cols)

    const int blockM = blockIdx.x * 64;
    const int blockN = blockIdx.y * 256;

    v8f acc[2][4] = {};

    auto stage = [&](int k0, int b) {
        {   // A tile 64x64 -> Ash[row][k]
            const int row = tid >> 2;                // 0..63
            const int kk  = (tid & 3) * 16;          // 0,16,32,48
            if constexpr (ASYNC_A) {
                const _Float16* g = A + (size_t)(blockM + row) * K + k0 + kk;
                const unsigned l = lds_off(&Ash[b][row * LDSSTRIDE + kk]);
                async_b128(l,      g);
                async_b128(l + 16, g + 8);
            } else {
                const AT* g = A + (size_t)(blockM + row) * K + k0 + kk;
                if (k0 + KTILE < K)
                    __builtin_prefetch((const void*)(g + KTILE), 0, 1);
#pragma unroll
                for (int i = 0; i < 16; ++i)
                    Ash[b][row * LDSSTRIDE + kk + i] = (_Float16)(float)g[i];
            }
        }
        if constexpr (ASYNC_B) {
            // B rows already [N][K]: 64 halfs = 8 x b128 per thread/row
            const _Float16* g = B + (size_t)(blockN + tid) * K + k0;
            const unsigned l = lds_off(&Bsh[b][tid * LDSSTRIDE]);
#pragma unroll
            for (int i = 0; i < 8; ++i)
                async_b128(l + 16 * i, g + 8 * i);
        } else if constexpr (B_NXK) {
            const BT* g = B + (size_t)(blockN + tid) * K + k0;
#pragma unroll
            for (int i = 0; i < KTILE; ++i)
                Bsh[b][tid * LDSSTRIDE + i] = (_Float16)(float)g[i];
        } else {
            // B is [K][N]; coalesced along N, transposed into Bsh[n][k]
#pragma unroll
            for (int i = 0; i < KTILE; ++i) {
                const int linear = tid + 256 * i;
                const int kk = linear >> 8;          // 0..63
                const int n  = linear & 255;
                Bsh[b][n * LDSSTRIDE + kk] =
                    (_Float16)(float)B[(size_t)(k0 + kk) * N + blockN + n];
            }
        }
    };

    stage(0, 0);
    int cur = 0;
    for (int k0 = 0; k0 < K; k0 += KTILE) {
        if constexpr (ASYNC_A || ASYNC_B)
            asm volatile("s_wait_asynccnt 0x0" ::: "memory");
        __syncthreads();                  // staging of buf[cur] complete
        if (k0 + KTILE < K) stage(k0 + KTILE, cur ^ 1);  // overlap with WMMAs

        const _Float16* As = Ash[cur];
        const _Float16* Bs = Bsh[cur];
#pragma unroll
        for (int ks = 0; ks < KTILE; ks += 32) {
            v16h aF[2], bF[4];
#pragma unroll
            for (int mi = 0; mi < 2; ++mi)
                aF[mi] = load_frag_nk(As + ks, wm * 32 + mi * 16, lane);
#pragma unroll
            for (int ni = 0; ni < 4; ++ni)
                bF[ni] = load_frag_nk(Bs + ks, wn * 64 + ni * 16, lane);
#pragma unroll
            for (int mi = 0; mi < 2; ++mi)
#pragma unroll
                for (int ni = 0; ni < 4; ++ni)
                    acc[mi][ni] = __builtin_amdgcn_wmma_f32_16x16x32_f16(
                        false, aF[mi], false, bF[ni],
                        (short)0, acc[mi][ni], false, false);
        }
        cur ^= 1;
    }

    // Store C (f32 C/D layout: VGPR v -> M=v (+8 upper half-wave), N=lane&15)
    const int cRow0 = blockM + wm * 32 + ((lane >> 4) * 8);
    const int cCol0 = blockN + wn * 64 + (lane & 15);
#pragma unroll
    for (int mi = 0; mi < 2; ++mi)
#pragma unroll
        for (int ni = 0; ni < 4; ++ni)
#pragma unroll
            for (int v = 0; v < 8; ++v) {
                const int r = cRow0 + mi * 16 + v;
                const int c = cCol0 + ni * 16;
                if constexpr (STORE_T)
                    C[(size_t)c * M + r] = (OT)(acc[mi][ni][v] * scale);
                else
                    C[(size_t)r * N + c] = (OT)(acc[mi][ni][v] * scale);
            }
}

// Row-wise softmax in place over f16 matrix P[rows][N], one block per row.
__global__ __launch_bounds__(256) void softmax_rows_kernel(
    _Float16* __restrict__ P, int N)
{
    __shared__ float    red[8];
    __shared__ _Float16 rowbuf[S_LEN];

    const int row = blockIdx.x;
    const int tid = threadIdx.x;
    _Float16* prow = P + (size_t)row * N;

    float vmax = -3.0e38f;
    for (int i = tid; i < N; i += 256) {
        const _Float16 h = prow[i];
        rowbuf[i] = h;
        vmax = fmaxf(vmax, (float)h);
    }
#pragma unroll
    for (int off = 16; off > 0; off >>= 1)
        vmax = fmaxf(vmax, __shfl_xor(vmax, off, 32));
    if ((tid & 31) == 0) red[tid >> 5] = vmax;
    __syncthreads();
    float m = red[0];
#pragma unroll
    for (int w = 1; w < 8; ++w) m = fmaxf(m, red[w]);
    __syncthreads();

    float vsum = 0.f;
    for (int i = tid; i < N; i += 256)
        vsum += __expf((float)rowbuf[i] - m);
#pragma unroll
    for (int off = 16; off > 0; off >>= 1)
        vsum += __shfl_xor(vsum, off, 32);
    if ((tid & 31) == 0) red[tid >> 5] = vsum;
    __syncthreads();
    float s = 0.f;
#pragma unroll
    for (int w = 0; w < 8; ++w) s += red[w];
    const float inv = 1.0f / s;

    for (int i = tid; i < N; i += 256)
        prow[i] = (_Float16)(__expf((float)rowbuf[i] - m) * inv);
}

extern "C" void kernel_launch(void* const* d_in, const int* in_sizes, int n_in,
                              void* d_out, int out_size, void* d_ws, size_t ws_size,
                              hipStream_t stream) {
    const float* q  = (const float*)d_in[0];
    const float* k  = (const float*)d_in[1];
    const float* v  = (const float*)d_in[2];
    const float* wQ = (const float*)d_in[3];
    const float* wK = (const float*)d_in[4];
    const float* wV = (const float*)d_in[5];
    float* out = (float*)d_out;

    // workspace layout (f16): Qh | Kh | Vt | P   (16+16+16+128 MB)
    _Float16* Qh = (_Float16*)d_ws;
    _Float16* Kh = Qh + (size_t)S_LEN * DK;
    _Float16* Vt = Kh + (size_t)S_LEN * DK;   // stored [DV][S]
    _Float16* P  = Vt + (size_t)S_LEN * DV;

    const dim3 blk(256);

    // 1) projections: [S,1024] x [1024,1024] -> f16 (V stored transposed)
    const dim3 gProj(S_LEN / 64, DK / 256);
    gemm_wmma_kernel<float, float, false, _Float16, false>
        <<<gProj, blk, 0, stream>>>(q, wQ, Qh, S_LEN, DK, D_INF, 1.0f);
    gemm_wmma_kernel<float, float, false, _Float16, false>
        <<<gProj, blk, 0, stream>>>(k, wK, Kh, S_LEN, DK, D_INF, 1.0f);
    gemm_wmma_kernel<float, float, false, _Float16, true>
        <<<gProj, blk, 0, stream>>>(v, wV, Vt, S_LEN, DV, D_INF, 1.0f);

    // 2) scores: P = f16(Q K^T / 64)  (async staging both operands)
    const dim3 gScore(S_LEN / 64, S_LEN / 256);
    gemm_wmma_kernel<_Float16, _Float16, true, _Float16, false>
        <<<gScore, blk, 0, stream>>>(Qh, Kh, P, S_LEN, S_LEN, DK, 1.0f / 64.0f);

    // 3) row softmax in place
    softmax_rows_kernel<<<dim3(S_LEN), blk, 0, stream>>>(P, S_LEN);

    // 4) out = P @ V = P x Vt^T -> f32  (async staging both operands)
    const dim3 gOut(S_LEN / 64, DV / 256);
    gemm_wmma_kernel<_Float16, _Float16, true, float, false>
        <<<gOut, blk, 0, stream>>>(P, Vt, out, S_LEN, DV, S_LEN, 1.0f);
}
